// OFTA_40913858461918
// MI455X (gfx1250) — matrix operand order repaired and checked by reading, not compiled, not run on hardware
//
#include <hip/hip_runtime.h>
#include <hip/hip_bf16.h>
#include <math.h>

// ---------------------------------------------------------------------------
// Problem constants (from reference): B=32, C=384, H=W=48 -> N=2304,
// HID=512, D=128, K=64, 3 Sinkhorn iters.
// ---------------------------------------------------------------------------
#define BATCH   32
#define C_IN    384
#define N_TOK   2304
#define HID     512
#define D_OUT   128
#define K_CL    64
#define MCAT    1024          // [w1; s1] rows
#define M2CAT   192           // [w2; s2] rows

typedef __bf16 bf16;
typedef __attribute__((ext_vector_type(16))) __bf16 v16bf;
typedef __attribute__((ext_vector_type(8)))  __bf16 v8bf;
typedef __attribute__((ext_vector_type(8)))  float  v8f;

union FragA { v16bf v; v8bf h[2]; };

__device__ __forceinline__ v8f wmma_bf16(v16bf a, v16bf b, v8f c) {
    // D = A(16x32 bf16) * B(32x16 bf16) + C(16x16 f32)
    return __builtin_amdgcn_wmma_f32_16x16x32_bf16(
        /*neg_a=*/false, a, /*neg_b=*/false, b,
        /*c_mod=*/(short)0, c, /*reuse_a=*/false, /*reuse_b=*/false);
}

// ---------------------------------------------------------------------------
// Kernel 0a: x fp32 -> bf16
// ---------------------------------------------------------------------------
__global__ __launch_bounds__(256) void conv_x_kernel(const float* __restrict__ x,
                                                     bf16* __restrict__ xb, int total) {
    int i = blockIdx.x * 256 + threadIdx.x;
    if (i < total) xb[i] = (bf16)x[i];
}

// ---------------------------------------------------------------------------
// Kernel 0b: pack [w1;s1]->W1c bf16, [b1;c1]->B1c, [w2;s2]->W2c bf16, [b2;c2]->B2c
// ---------------------------------------------------------------------------
__global__ __launch_bounds__(256) void pack_w_kernel(
    const float* __restrict__ w1, const float* __restrict__ b1,
    const float* __restrict__ w2, const float* __restrict__ b2,
    const float* __restrict__ s1, const float* __restrict__ c1,
    const float* __restrict__ s2, const float* __restrict__ c2,
    bf16* __restrict__ W1c, float* __restrict__ B1c,
    bf16* __restrict__ W2c, float* __restrict__ B2c) {
    int i = blockIdx.x * 256 + threadIdx.x;
    const int W1H = HID * C_IN;          // 196608
    if (i < W1H) { W1c[i] = (bf16)w1[i]; W1c[W1H + i] = (bf16)s1[i]; }
    if (i < D_OUT * HID) W2c[i] = (bf16)w2[i];                  // 65536
    if (i < K_CL * HID)  W2c[D_OUT * HID + i] = (bf16)s2[i];    // 32768
    if (i < HID) { B1c[i] = b1[i]; B1c[HID + i] = c1[i]; }
    if (i < D_OUT) B2c[i] = b2[i];
    if (i < K_CL)  B2c[D_OUT + i] = c2[i];
}

// ---------------------------------------------------------------------------
// Kernel 1: Hcat^T[b][n][m] = relu( W1c[m,:] . Xb[b][:,n] + B1c[m] )  (bf16)
//   A = W1c [1024 x 384] row-major (direct global fragment loads)
//   B = X   [384 x 2304] staged transposed through LDS (32K x 32N tile)
//   Register blocking: each wave owns a 32(M) x 32(N) tile = 2x2 WMMA tiles,
//   so every fragment feeds 2 WMMAs and each staging barrier feeds 4 WMMAs.
//   Block = 128 threads (4 waves stacked in M): block tile 128(M) x 32(N).
//   grid (72, 8, 32)
// ---------------------------------------------------------------------------
__global__ __launch_bounds__(128) void gemm1_kernel(
    const bf16* __restrict__ Xbf, const bf16* __restrict__ W1c,
    const float* __restrict__ B1c, bf16* __restrict__ HT) {
    const int b    = blockIdx.z;
    const int m0   = blockIdx.y * 128;
    const int n0   = blockIdx.x * 32;
    const int tid  = threadIdx.x;
    const int wave = tid >> 5;
    const int lane = tid & 31;
    const int lhi  = lane >> 4;
    const int llo  = lane & 15;

    __shared__ __align__(32) __bf16 ldsX[32][48];   // [n][k], padded rows

    const bf16* Xb    = Xbf + (size_t)b * C_IN * N_TOK;
    const int   mW    = m0 + wave * 32;
    const bf16* Arow0 = W1c + (size_t)(mW + llo) * C_IN;
    const bf16* Arow1 = W1c + (size_t)(mW + 16 + llo) * C_IN;

    v8f acc00 = {}, acc01 = {}, acc10 = {}, acc11 = {};
    for (int k0 = 0; k0 < C_IN; k0 += 32) {
        // stage X[k0..k0+32) x [n0..n0+32) transposed into LDS
        {
            const int kk = tid >> 2;            // 0..31
            const int nn = (tid & 3) * 8;       // 0,8,16,24
            v8bf t = *(const v8bf*)(Xb + (size_t)(k0 + kk) * N_TOK + n0 + nn);
#pragma unroll
            for (int i = 0; i < 8; ++i) ldsX[nn + i][kk] = t[i];
        }
        __syncthreads();
        FragA fa0, fa1;
        fa0.h[0] = *(const v8bf*)(Arow0 + k0 + 16 * lhi);
        fa0.h[1] = *(const v8bf*)(Arow0 + k0 + 16 * lhi + 8);
        fa1.h[0] = *(const v8bf*)(Arow1 + k0 + 16 * lhi);
        fa1.h[1] = *(const v8bf*)(Arow1 + k0 + 16 * lhi + 8);
        v16bf fb0 = *(const v16bf*)(&ldsX[llo][16 * lhi]);
        v16bf fb1 = *(const v16bf*)(&ldsX[16 + llo][16 * lhi]);
        acc00 = wmma_bf16(fa0.v, fb0, acc00);
        acc01 = wmma_bf16(fa0.v, fb1, acc01);
        acc10 = wmma_bf16(fa1.v, fb0, acc10);
        acc11 = wmma_bf16(fa1.v, fb1, acc11);
        __syncthreads();
    }
    // epilogue: bias + relu, store token-major (one b128 per lane per tile)
    v8f accs[2][2] = { { acc00, acc01 }, { acc10, acc11 } };
#pragma unroll
    for (int mi = 0; mi < 2; ++mi) {
        const int mBase = mW + mi * 16 + 8 * lhi;
#pragma unroll
        for (int ni = 0; ni < 2; ++ni) {
            const int n = n0 + ni * 16 + llo;
            v8bf outv;
#pragma unroll
            for (int j = 0; j < 8; ++j) {
                float vo = accs[mi][ni][j] + B1c[mBase + j];
                outv[j]  = (bf16)fmaxf(vo, 0.0f);
            }
            *(v8bf*)(HT + ((size_t)b * N_TOK + n) * MCAT + mBase) = outv;
        }
    }
}

// ---------------------------------------------------------------------------
// Kernel 2: rows 0..127 -> F bf16 [b][128][2304]; rows 128..191 -> Zp f32 [b][64][2304]
//   A = W2c [192 x 512] row-major, B = Hcat^T (contiguous along hid) -- no LDS
//   Register blocking: wave tile 16(M) x 32(N), A fragment reused by 2 WMMAs.
//   grid (72, 3, 32), block 128 (4 waves stacked in M -> block tile 64 x 32)
// ---------------------------------------------------------------------------
__global__ __launch_bounds__(128) void gemm2_kernel(
    const bf16* __restrict__ HT, const bf16* __restrict__ W2c,
    const float* __restrict__ B2c, bf16* __restrict__ Fbf, float* __restrict__ Zp) {
    const int b    = blockIdx.z;
    const int m0   = blockIdx.y * 64;
    const int n0   = blockIdx.x * 32;
    const int wave = threadIdx.x >> 5;
    const int lane = threadIdx.x & 31;
    const int lhi  = lane >> 4;
    const int llo  = lane & 15;

    const int   mW     = m0 + wave * 16;
    const int   colOff = (m0 >= D_OUT) ? HID : 0;      // f uses h, p uses hs
    const bf16* Arow   = W2c + (size_t)(mW + llo) * HID;
    const bf16* Brow0  = HT + ((size_t)b * N_TOK + n0 + llo) * MCAT + colOff;
    const bf16* Brow1  = HT + ((size_t)b * N_TOK + n0 + 16 + llo) * MCAT + colOff;

    v8f acc0 = {}, acc1 = {};
#pragma unroll 4
    for (int k0 = 0; k0 < HID; k0 += 32) {
        FragA fa;
        fa.h[0] = *(const v8bf*)(Arow + k0 + 16 * lhi);
        fa.h[1] = *(const v8bf*)(Arow + k0 + 16 * lhi + 8);
        v16bf fb0 = *(const v16bf*)(Brow0 + k0 + 16 * lhi);
        v16bf fb1 = *(const v16bf*)(Brow1 + k0 + 16 * lhi);
        acc0 = wmma_bf16(fa.v, fb0, acc0);
        acc1 = wmma_bf16(fa.v, fb1, acc1);
    }
    const int mBase = mW + 8 * lhi;
    v8f accs[2] = { acc0, acc1 };
#pragma unroll
    for (int ni = 0; ni < 2; ++ni) {
        const int n = n0 + ni * 16 + llo;
        if (m0 < D_OUT) {
#pragma unroll
            for (int j = 0; j < 8; ++j) {
                float vo = accs[ni][j] + B2c[mBase + j];
                Fbf[((size_t)b * D_OUT + mBase + j) * N_TOK + n] = (bf16)vo;
            }
        } else {
#pragma unroll
            for (int j = 0; j < 8; ++j) {
                float vo = accs[ni][j] + B2c[mBase + j];
                Zp[((size_t)b * K_CL + (mBase + j - D_OUT)) * N_TOK + n] = vo;
            }
        }
    }
}

// ---------------------------------------------------------------------------
// Kernel 3: log-domain Sinkhorn (3 iters) + P = exp(Z+u+v-norm) in bf16.
// Dust-bin row (value alpha) is handled in registers, never stored.
// One block (1024 threads = 32 waves) per batch.
// ---------------------------------------------------------------------------
__global__ __launch_bounds__(1024) void sinkhorn_kernel(
    const float* __restrict__ Zp, const float* __restrict__ alphaPtr,
    bf16* __restrict__ Pbf) {
    const int b = blockIdx.x;
    const float* Z = Zp + (size_t)b * K_CL * N_TOK;
    const float alpha = alphaPtr[0];

    __shared__ float u_s[K_CL + 1];
    __shared__ float v_s[N_TOK];

    const int tid  = threadIdx.x;
    const int lane = tid & 31;
    const int wave = tid >> 5;

    const float norm    = -__logf((float)(K_CL + N_TOK));           // -log(2368)
    const float lmu_bin = __logf((float)(N_TOK - K_CL)) + norm;     //  log(2240)+norm

    for (int n = tid; n < N_TOK; n += 1024) v_s[n] = 0.0f;
    __syncthreads();

    for (int it = 0; it < 3; ++it) {
        // u[m] = log_mu[m] - logsumexp_n(Z[m,n] + v[n]);  one wave per row
        for (int m = wave; m < K_CL + 1; m += 32) {
            float mx = -3.0e38f, sm = 0.0f;
            if (m < K_CL) {
                const float* zr = Z + (size_t)m * N_TOK;
                for (int n = lane; n < N_TOK; n += 32) {
                    float t  = zr[n] + v_s[n];
                    float nm = fmaxf(mx, t);
                    sm = sm * __expf(mx - nm) + __expf(t - nm);
                    mx = nm;
                }
            } else {
                for (int n = lane; n < N_TOK; n += 32) {
                    float t  = alpha + v_s[n];
                    float nm = fmaxf(mx, t);
                    sm = sm * __expf(mx - nm) + __expf(t - nm);
                    mx = nm;
                }
            }
#pragma unroll
            for (int off = 16; off >= 1; off >>= 1) {   // wave32 reduction
                float omx = __shfl_xor(mx, off, 32);
                float osm = __shfl_xor(sm, off, 32);
                float nm  = fmaxf(mx, omx);
                sm = sm * __expf(mx - nm) + osm * __expf(omx - nm);
                mx = nm;
            }
            if (lane == 0) {
                float lmu = (m < K_CL) ? norm : lmu_bin;
                u_s[m] = lmu - (mx + __logf(sm));
            }
        }
        __syncthreads();
        // v[n] = log_nu - logsumexp_m(Z[m,n] + u[m]);  one thread per column
        for (int n = tid; n < N_TOK; n += 1024) {
            float mx = -3.0e38f, sm = 0.0f;
            for (int m = 0; m < K_CL; ++m) {
                float t  = Z[(size_t)m * N_TOK + n] + u_s[m];
                float nm = fmaxf(mx, t);
                sm = sm * __expf(mx - nm) + __expf(t - nm);
                mx = nm;
            }
            {   // dust-bin row
                float t  = alpha + u_s[K_CL];
                float nm = fmaxf(mx, t);
                sm = sm * __expf(mx - nm) + __expf(t - nm);
                mx = nm;
            }
            v_s[n] = norm - (mx + __logf(sm));
        }
        __syncthreads();
    }
    // P (drop dust bin) -> bf16, row-major [64, 2304] (perfect agg B operand)
    for (int idx = tid; idx < K_CL * N_TOK; idx += 1024) {
        int m = idx / N_TOK;
        int n = idx - m * N_TOK;
        float e = __expf(Z[idx] + u_s[m] + v_s[n] - norm);
        Pbf[(size_t)b * K_CL * N_TOK + idx] = (bf16)e;
    }
}

// ---------------------------------------------------------------------------
// Kernel 4: agg[b,d,k] = sum_n F[b,d,n] * P[b,k,n]   (M=128, N=64, K=2304)
//   A = F row-major, B = P row-major (col-of-B contiguous along n) -- no LDS
//   Register blocking: wave tile 16(M) x 32(N), A fragment reused by 2 WMMAs.
//   grid (2, 32), block 256 (8 waves cover full M=128)
// ---------------------------------------------------------------------------
__global__ __launch_bounds__(256) void agg_kernel(
    const bf16* __restrict__ Fbf, const bf16* __restrict__ Pbf,
    float* __restrict__ out) {
    const int b    = blockIdx.y;
    const int n0   = blockIdx.x * 32;
    const int wave = threadIdx.x >> 5;
    const int lane = threadIdx.x & 31;
    const int lhi  = lane >> 4;
    const int llo  = lane & 15;

    const int   mW    = wave * 16;
    const bf16* Arow  = Fbf + ((size_t)b * D_OUT + mW + llo) * N_TOK;
    const bf16* Bcol0 = Pbf + ((size_t)b * K_CL + n0 + llo) * N_TOK;
    const bf16* Bcol1 = Pbf + ((size_t)b * K_CL + n0 + 16 + llo) * N_TOK;

    v8f acc0 = {}, acc1 = {};
#pragma unroll 4
    for (int k0 = 0; k0 < N_TOK; k0 += 32) {
        FragA fa;
        fa.h[0] = *(const v8bf*)(Arow + k0 + 16 * lhi);
        fa.h[1] = *(const v8bf*)(Arow + k0 + 16 * lhi + 8);
        v16bf fb0 = *(const v16bf*)(Bcol0 + k0 + 16 * lhi);
        v16bf fb1 = *(const v16bf*)(Bcol1 + k0 + 16 * lhi);
        acc0 = wmma_bf16(fa.v, fb0, acc0);
        acc1 = wmma_bf16(fa.v, fb1, acc1);
    }
    const int mBase = mW + 8 * lhi;
    v8f accs[2] = { acc0, acc1 };
#pragma unroll
    for (int ni = 0; ni < 2; ++ni) {
        const int cl = n0 + ni * 16 + llo;
#pragma unroll
        for (int j = 0; j < 8; ++j)
            out[(size_t)b * (D_OUT * K_CL) + (size_t)(mBase + j) * K_CL + cl] = accs[ni][j];
    }
}

// ---------------------------------------------------------------------------
// Kernel 5: in-place L2 normalization over d for each (b, k)
// ---------------------------------------------------------------------------
__global__ __launch_bounds__(64) void norm_kernel(float* __restrict__ out) {
    const int b = blockIdx.x;
    const int k = threadIdx.x;
    float* p = out + (size_t)b * (D_OUT * K_CL);
    float ss = 0.0f;
    for (int d = 0; d < D_OUT; ++d) { float v = p[(size_t)d * K_CL + k]; ss += v * v; }
    float scale = 1.0f / fmaxf(sqrtf(ss), 1e-12f);
    for (int d = 0; d < D_OUT; ++d) p[(size_t)d * K_CL + k] *= scale;
}

// ---------------------------------------------------------------------------
extern "C" void kernel_launch(void* const* d_in, const int* in_sizes, int n_in,
                              void* d_out, int out_size, void* d_ws, size_t ws_size,
                              hipStream_t stream) {
    const float* x  = (const float*)d_in[0];
    const float* w1 = (const float*)d_in[1];
    const float* b1 = (const float*)d_in[2];
    const float* w2 = (const float*)d_in[3];
    const float* b2 = (const float*)d_in[4];
    const float* s1 = (const float*)d_in[5];
    const float* c1 = (const float*)d_in[6];
    const float* s2 = (const float*)d_in[7];
    const float* c2 = (const float*)d_in[8];
    const float* al = (const float*)d_in[9];
    float* out = (float*)d_out;
    (void)in_sizes; (void)n_in; (void)out_size; (void)ws_size;

    size_t off = 0;
    auto take = [&](size_t nbytes) -> void* {
        void* p = (char*)d_ws + off;
        off += (nbytes + 255) & ~(size_t)255;
        return p;
    };
    const size_t XTOT = (size_t)BATCH * C_IN * N_TOK;              // 28,311,552
    bf16*  Xbf = (bf16*) take(XTOT * sizeof(bf16));
    bf16*  HT  = (bf16*) take((size_t)BATCH * N_TOK * MCAT * sizeof(bf16));
    bf16*  Fbf = (bf16*) take((size_t)BATCH * D_OUT * N_TOK * sizeof(bf16));
    float* Zp  = (float*)take((size_t)BATCH * K_CL * N_TOK * sizeof(float));
    bf16*  Pbf = (bf16*) take((size_t)BATCH * K_CL * N_TOK * sizeof(bf16));
    bf16*  W1c = (bf16*) take((size_t)MCAT * C_IN * sizeof(bf16));
    float* B1c = (float*)take(MCAT * sizeof(float));
    bf16*  W2c = (bf16*) take((size_t)M2CAT * HID * sizeof(bf16));
    float* B2c = (float*)take(M2CAT * sizeof(float));

    conv_x_kernel<<<(unsigned)((XTOT + 255) / 256), 256, 0, stream>>>(x, Xbf, (int)XTOT);
    pack_w_kernel<<<(HID * C_IN + 255) / 256, 256, 0, stream>>>(
        w1, b1, w2, b2, s1, c1, s2, c2, W1c, B1c, W2c, B2c);

    gemm1_kernel<<<dim3(N_TOK / 32, MCAT / 128, BATCH), 128, 0, stream>>>(Xbf, W1c, B1c, HT);
    gemm2_kernel<<<dim3(N_TOK / 32, M2CAT / 64, BATCH), 128, 0, stream>>>(HT, W2c, B2c, Fbf, Zp);
    sinkhorn_kernel<<<BATCH, 1024, 0, stream>>>(Zp, al, Pbf);
    agg_kernel<<<dim3(K_CL / 32, BATCH), 256, 0, stream>>>(Fbf, Pbf, out);
    norm_kernel<<<BATCH, K_CL, 0, stream>>>(out);
}
